// APCNNCropMine_50740743635441
// MI455X (gfx1250) — compile-verified, matrix operand order checked by reading, not run on hardware
//
#include <hip/hip_runtime.h>
#include <hip/hip_bf16.h>
#include <climits>

// ---------------- problem constants ----------------
#define NIMG 16
#define NCH  256
// level geometry: lvl0 -> x3 (56x56, stride 8,  s 32)
//                 lvl1 -> x4 (28x28, stride 16, s 64)
//                 lvl2 -> x5 (14x14, stride 32, s 128)
constexpr int HW3 = 56 * 56, HW4 = 28 * 28, HW5 = 14 * 14;

// ---------------- workspace layout (float/int indices, 4B units) ----------------
constexpr size_t OFF_M3   = 0;
constexpr size_t OFF_M4   = OFF_M3 + (size_t)NIMG * HW3;          // 50176
constexpr size_t OFF_M5   = OFF_M4 + (size_t)NIMG * HW4;          // 62720
constexpr size_t OFF_SUM  = OFF_M5 + (size_t)NIMG * HW5;          // 65856 : 48 floats [lvl*16+i]
constexpr size_t OFF_ROI  = OFF_SUM + 48;                         // 16*3*5*6 floats (b0..b3, sc, pad)
constexpr size_t OFF_CNT  = OFF_ROI + (size_t)NIMG * 3 * 5 * 6;   // 48 ints
constexpr size_t OFF_RECT = OFF_CNT + 48;                         // 16*4 ints (x,y,w,h) for x2 crop
constexpr size_t OFF_REG  = OFF_RECT + 64;                        // 16*4*4 ints (xa,ya,xb,yb) x5 crops
constexpr size_t OFF_NCR  = OFF_REG + (size_t)NIMG * 4 * 4;       // 16 ints
constexpr size_t WS_END   = OFF_NCR + NIMG;

// ---------------- WMMA types ----------------
typedef float    v2f  __attribute__((ext_vector_type(2)));
typedef float    v8f  __attribute__((ext_vector_type(8)));
typedef _Float16 v16h __attribute__((ext_vector_type(16)));

__device__ inline size_t roi_slot(int i, int lvl, int k) {
    return OFF_ROI + ((size_t)(i * 3 + lvl) * 5 + k) * 6;
}

// ---------------- K0: zero the control region of ws ----------------
__global__ void zero_ws_kernel(float* p, int n) {
    int j = blockIdx.x * 256 + threadIdx.x;
    if (j < n) p[j] = 0.0f;
}

// ---------------- K1: channel mean via WMMA ----------------
// mask(1 x HW) = ones(1 x 256) * X(256 x HW) / 256, computed 16 columns per wave.
// A = all-ones => D[m][n] = sum_k B[k][n]; only N(lane)=lane&15 matters.
__global__ __launch_bounds__(32)
void channel_mean_wmma_kernel(const float* __restrict__ x, float* __restrict__ mask,
                              float* __restrict__ sum_lvl, int H, int W,
                              int wy0, int wy1, int wx0, int wx1) {
    const int HW   = H * W;
    const int img  = blockIdx.y;
    const int lane = threadIdx.x;
    const int col  = lane & 15;
    const int pos  = blockIdx.x * 16 + col;
    const int posc = pos < HW ? pos : HW - 1;
    const float* xi = x + (size_t)img * NCH * HW;

    v8f acc = {};
#if __has_builtin(__builtin_amdgcn_wmma_f32_16x16x4_f32)
    const int kadd = (lane >> 4) * 2;         // lanes 0-15 supply K pair {0,1}; 16-31 supply {2,3}
    v2f a; a.x = 1.0f; a.y = 1.0f;            // ones A-matrix (16x4)
    for (int cb = 0; cb < NCH; cb += 4) {
        v2f b;
        b.x = xi[(size_t)(cb + kadd + 0) * HW + posc];
        b.y = xi[(size_t)(cb + kadd + 1) * HW + posc];
        acc = __builtin_amdgcn_wmma_f32_16x16x4_f32(false, a, false, b,
                                                    (short)0, acc, false, false);
    }
#else
    // fallback: f16 WMMA 16x16x32 (codegen-confirmed builtin), f32 accumulate
    const int kadd = (lane >> 4) * 16;
    v16h a;
    #pragma unroll
    for (int t = 0; t < 16; t++) a[t] = (_Float16)1.0f;
    for (int cb = 0; cb < NCH; cb += 32) {
        v16h b;
        #pragma unroll
        for (int t = 0; t < 16; t++)
            b[t] = (_Float16)xi[(size_t)(cb + kadd + t) * HW + posc];
        acc = __builtin_amdgcn_wmma_f32_16x16x32_f16(false, a, false, b,
                                                     (short)0, acc, false, false);
    }
#endif

    // D row 0 lives in acc[0] of lanes 0..15 (all rows identical since A==1)
    __shared__ float sbuf[16];
    if (lane < 16) {
        float v = 0.0f;
        if (pos < HW) {
            float m  = acc[0] * (1.0f / (float)NCH);
            int   py = pos / W, px = pos % W;
            bool inwin = (py >= wy0) && (py < wy1) && (px >= wx0) && (px < wx1);
            v = inwin ? m : 0.0f;
            mask[(size_t)img * HW + pos] = v;   // windowed mask (zeros outside)
        }
        sbuf[lane] = v;
    }
    __syncthreads();
    if (lane == 0) {
        float s = 0.0f;
        #pragma unroll
        for (int t = 0; t < 16; t++) s += sbuf[t];
        atomicAdd(&sum_lvl[img], s);
    }
}

// ---------------- K2: per-(image,level) NMS (topk=5, iou 0.05) ----------------
__global__ __launch_bounds__(256)
void nms_level_kernel(float* ws) {
    __shared__ float         sc[HW3];
    __shared__ unsigned char alive[HW3];
    __shared__ float         rv[256];
    __shared__ int           ri[256];
    __shared__ int           s_chosen;

    const int i = blockIdx.x, lvl = blockIdx.y, tid = threadIdx.x;
    int HW, W; float stride, s; size_t moff;
    if (lvl == 0)      { HW = HW3; W = 56; stride =  8.f; s =  32.f; moff = OFF_M3; }
    else if (lvl == 1) { HW = HW4; W = 28; stride = 16.f; s =  64.f; moff = OFF_M4; }
    else               { HW = HW5; W = 14; stride = 32.f; s = 128.f; moff = OFF_M5; }

    const float* mask = ws + moff + (size_t)i * HW;
    const float  mean = ws[OFF_SUM + lvl * NIMG + i] / (float)HW;

    for (int j = tid; j < HW; j += 256) {
        float v = mask[j];
        sc[j]    = v;
        alive[j] = (v > mean) ? 1 : 0;
    }
    __syncthreads();

    const float halfs = 0.5f * s;
    const float area2 = 2.0f * s * s;   // area_i + area_b for equal unclipped squares
    int count = 0;

    for (int k = 0; k < 5; k++) {
        float bestv = -__FLT_MAX__; int besti = INT_MAX;
        for (int j = tid; j < HW; j += 256)
            if (alive[j]) {
                float v = sc[j];
                if (v > bestv || (v == bestv && j < besti)) { bestv = v; besti = j; }
            }
        rv[tid] = bestv; ri[tid] = besti;
        __syncthreads();
        for (int off = 128; off > 0; off >>= 1) {
            if (tid < off) {
                float v2 = rv[tid + off]; int i2 = ri[tid + off];
                if (v2 > rv[tid] || (v2 == rv[tid] && i2 < ri[tid])) { rv[tid] = v2; ri[tid] = i2; }
            }
            __syncthreads();
        }
        if (tid == 0) s_chosen = (rv[0] > -__FLT_MAX__) ? ri[0] : -1;
        __syncthreads();
        const int ch = s_chosen;
        if (ch < 0) break;

        const float cxc = (float)(ch % W) * stride, cyc = (float)(ch / W) * stride;
        const float cb0 = cxc - halfs, cb1 = cyc - halfs, cb2 = cxc + halfs, cb3 = cyc + halfs;

        for (int j = tid; j < HW; j += 256)
            if (alive[j]) {
                float bx = (float)(j % W) * stride, by = (float)(j / W) * stride;
                float xx1 = fmaxf(cb0, bx - halfs), yy1 = fmaxf(cb1, by - halfs);
                float xx2 = fminf(cb2, bx + halfs), yy2 = fminf(cb3, by + halfs);
                float inter = fmaxf(xx2 - xx1, 0.f) * fmaxf(yy2 - yy1, 0.f);
                if (inter / (area2 - inter) > 0.05f) alive[j] = 0;  // also kills j==ch (iou=1)
            }
        if (tid == 0) {
            alive[ch] = 0;
            size_t ro = roi_slot(i, lvl, k);
            ws[ro + 0] = fmaxf(cb0, 0.f);
            ws[ro + 1] = fmaxf(cb1, 0.f);
            ws[ro + 2] = fminf(cb2, 447.f);
            ws[ro + 3] = fminf(cb3, 447.f);
            ws[ro + 4] = sc[ch];
        }
        count++;
        __syncthreads();
    }
    if (tid == 0) ((int*)ws)[OFF_CNT + i * 3 + lvl] = count;
}

// ---------------- K3a: per-image union crop rect on x2 (/8, int-trunc) ----------------
__global__ void union_rect_kernel(float* ws) {
    int i = threadIdx.x;
    if (i >= NIMG) return;
    int* wsi = (int*)ws;
    float mnx = 1e30f, mny = 1e30f, mxx = -1e30f, mxy = -1e30f;
    for (int lvl = 0; lvl < 3; lvl++) {
        int cnt = wsi[OFF_CNT + i * 3 + lvl];
        for (int k = 0; k < cnt; k++) {
            size_t ro = roi_slot(i, lvl, k);
            mnx = fminf(mnx, ws[ro + 0] * 0.125f);
            mny = fminf(mny, ws[ro + 1] * 0.125f);
            mxx = fmaxf(mxx, ws[ro + 2] * 0.125f);
            mxy = fmaxf(mxy, ws[ro + 3] * 0.125f);
        }
    }
    int x1c = (int)mnx, y1c = (int)mny, x2c = (int)mxx, y2c = (int)mxy;
    int w = x2c - x1c, h = y2c - y1c;
    if (w < 1) w = 1;
    if (h < 1) h = 1;
    if (x1c + w > 56) x1c = 56 - w;
    if (y1c + h > 56) y1c = 56 - h;
    if (x1c < 0) x1c = 0;
    if (y1c < 0) y1c = 0;
    wsi[OFF_RECT + i * 4 + 0] = x1c;
    wsi[OFF_RECT + i * 4 + 1] = y1c;
    wsi[OFF_RECT + i * 4 + 2] = w;
    wsi[OFF_RECT + i * 4 + 3] = h;
}

// ---------------- K3b: bilinear resize crop -> (256,56,56) ----------------
__global__ __launch_bounds__(256)
void resize_kernel(const float* __restrict__ x2, const float* __restrict__ ws,
                   float* __restrict__ out) {
    size_t idx = (size_t)blockIdx.x * 256 + threadIdx.x;
    const size_t total = (size_t)NIMG * NCH * 56 * 56;
    if (idx >= total) return;
    int ox = (int)(idx % 56);
    int oy = (int)((idx / 56) % 56);
    int c  = (int)((idx / 3136) % NCH);
    int i  = (int)(idx / ((size_t)3136 * NCH));

    const int* wsi = (const int*)ws;
    int rx = wsi[OFF_RECT + i * 4 + 0], ry = wsi[OFF_RECT + i * 4 + 1];
    int w  = wsi[OFF_RECT + i * 4 + 2], h  = wsi[OFF_RECT + i * 4 + 3];

    float sx = (ox + 0.5f) * (float)w * (1.f / 56.f) - 0.5f;
    float sy = (oy + 0.5f) * (float)h * (1.f / 56.f) - 0.5f;
    sx = fminf(fmaxf(sx, 0.f), (float)(w - 1));
    sy = fminf(fmaxf(sy, 0.f), (float)(h - 1));
    int x0 = (int)sx, y0 = (int)sy;
    int x1 = x0 + 1 < w ? x0 + 1 : w - 1;
    int y1 = y0 + 1 < h ? y0 + 1 : h - 1;
    float fx = sx - (float)x0, fy = sy - (float)y0;

    const float* p = x2 + ((size_t)i * NCH + c) * 3136;
    float v00 = p[(ry + y0) * 56 + rx + x0];
    float v01 = p[(ry + y0) * 56 + rx + x1];
    float v10 = p[(ry + y1) * 56 + rx + x0];
    float v11 = p[(ry + y1) * 56 + rx + x1];
    float v0 = v00 + (v01 - v00) * fx;
    float v1 = v10 + (v11 - v10) * fx;
    out[idx] = v0 + (v1 - v0) * fy;
}

// ---------------- K4a: second NMS (/32), one scalar thread per image ----------------
// Alive set is a bitmask; boxes re-read from ws (n <= 15) -> no local arrays,
// no shared memory, no divergent thread-0 blob (works around LTO backend crash).
__global__ void nms2_kernel(float* __restrict__ ws) {
    int i = blockIdx.x * blockDim.x + threadIdx.x;
    if (i >= NIMG) return;
    int* wsi = (int*)ws;

    const int c0 = wsi[OFF_CNT + i * 3 + 0];
    const int c1 = wsi[OFF_CNT + i * 3 + 1];
    const int c2 = wsi[OFF_CNT + i * 3 + 2];
    const int n  = c0 + c1 + c2;

    unsigned aliveMask = (n >= 32) ? 0xFFFFFFFFu : ((1u << n) - 1u);
    int ncr = 0;

    while (ncr < 4) {
        // argmax over alive rows (stable: strict > keeps lowest index on ties)
        int best = -1; float bv = -__FLT_MAX__;
        for (int r = 0; r < n; r++) {
            if (!((aliveMask >> r) & 1u)) continue;
            int lvl = (r < c0) ? 0 : ((r < c0 + c1) ? 1 : 2);
            int k   = (r < c0) ? r : ((r < c0 + c1) ? r - c0 : r - c0 - c1);
            float scv = ws[roi_slot(i, lvl, k) + 4];
            if (scv > bv) { bv = scv; best = r; }
        }
        if (best < 0) break;
        aliveMask &= ~(1u << best);

        int blvl = (best < c0) ? 0 : ((best < c0 + c1) ? 1 : 2);
        int bk   = (best < c0) ? best : ((best < c0 + c1) ? best - c0 : best - c0 - c1);
        size_t bo = roi_slot(i, blvl, bk);
        float a0 = ws[bo + 0] * (1.f / 32.f), a1 = ws[bo + 1] * (1.f / 32.f);
        float a2 = ws[bo + 2] * (1.f / 32.f), a3 = ws[bo + 3] * (1.f / 32.f);
        float areaA = (a2 - a0) * (a3 - a1);

        // suppress iou > 0.05
        for (int r = 0; r < n; r++) {
            if (!((aliveMask >> r) & 1u)) continue;
            int lvl = (r < c0) ? 0 : ((r < c0 + c1) ? 1 : 2);
            int k   = (r < c0) ? r : ((r < c0 + c1) ? r - c0 : r - c0 - c1);
            size_t ro = roi_slot(i, lvl, k);
            float b0 = ws[ro + 0] * (1.f / 32.f), b1 = ws[ro + 1] * (1.f / 32.f);
            float b2 = ws[ro + 2] * (1.f / 32.f), b3 = ws[ro + 3] * (1.f / 32.f);
            float xx1 = fmaxf(a0, b0), yy1 = fmaxf(a1, b1);
            float xx2 = fminf(a2, b2), yy2 = fminf(a3, b3);
            float inter = fmaxf(xx2 - xx1, 0.f) * fmaxf(yy2 - yy1, 0.f);
            float areaB = (b2 - b0) * (b3 - b1);
            if (inter / (areaA + areaB - inter) > 0.05f) aliveMask &= ~(1u << r);
        }

        // min-size filter and record crop region
        int xa = (int)floorf(a0), ya = (int)floorf(a1);
        int xb = (int)ceilf(a2),  yb = (int)ceilf(a3);
        if ((xb - xa) >= 2 || (yb - ya) >= 2) {
            if (xa < 0) xa = 0;
            if (ya < 0) ya = 0;
            if (xb > 14) xb = 14;
            if (yb > 14) yb = 14;
            size_t go = OFF_REG + ((size_t)i * 4 + ncr) * 4;
            wsi[go + 0] = xa; wsi[go + 1] = ya;
            wsi[go + 2] = xb; wsi[go + 3] = yb;
            ncr++;
        }
    }
    wsi[OFF_NCR + i] = ncr;
}

// ---------------- K4b: region means on x5 -> (16,4,256), straight-line ----------------
__global__ __launch_bounds__(256)
void crop_mean_kernel(const float* __restrict__ x5, const float* __restrict__ ws,
                      float* __restrict__ outc) {
    const int i = blockIdx.x;
    const int c = threadIdx.x;                 // 256 threads == 256 channels
    const int* wsi = (const int*)ws;
    const int ncr = wsi[OFF_NCR + i];
    const float* xi = x5 + ((size_t)i * NCH + c) * HW5;

    float last = 0.f;
    if (ncr == 0) {
        float sum = 0.f;
        for (int p = 0; p < HW5; p++) sum += xi[p];
        last = sum * (1.f / (float)HW5);
        outc[((size_t)i * 4 + 0) * NCH + c] = last;
    }
    for (int k = 0; k < ncr; k++) {
        size_t go = OFF_REG + ((size_t)i * 4 + k) * 4;
        int xa = wsi[go + 0], ya = wsi[go + 1], xb = wsi[go + 2], yb = wsi[go + 3];
        float sum = 0.f;
        int cnt = (xb - xa) * (yb - ya);
        for (int y = ya; y < yb; y++)
            for (int x = xa; x < xb; x++)
                sum += xi[y * 14 + x];
        last = (cnt > 0) ? sum / (float)cnt : 0.f;
        outc[((size_t)i * 4 + k) * NCH + c] = last;
    }
    int start = (ncr == 0) ? 1 : ncr;
    for (int k = start; k < 4; k++)
        outc[((size_t)i * 4 + k) * NCH + c] = last;
}

// ---------------- launch ----------------
extern "C" void kernel_launch(void* const* d_in, const int* in_sizes, int n_in,
                              void* d_out, int out_size, void* d_ws, size_t ws_size,
                              hipStream_t stream) {
    const float* x2 = (const float*)d_in[1];   // (16,256,56,56)
    const float* x3 = (const float*)d_in[2];   // (16,256,56,56)
    const float* x4 = (const float*)d_in[3];   // (16,256,28,28)
    const float* x5 = (const float*)d_in[4];   // (16,256,14,14)
    float* out = (float*)d_out;
    float* ws  = (float*)d_ws;

    // zero sums/roi/counts/rects/regions region
    {
        int n = (int)(WS_END - OFF_SUM);
        zero_ws_kernel<<<(n + 255) / 256, 256, 0, stream>>>(ws + OFF_SUM, n);
    }

    // channel means via WMMA (windows = int(0.1*h)..int(0.9*h))
    channel_mean_wmma_kernel<<<dim3((HW3 + 15) / 16, NIMG), 32, 0, stream>>>(
        x3, ws + OFF_M3, ws + OFF_SUM + 0 * NIMG, 56, 56, 5, 50, 5, 50);
    channel_mean_wmma_kernel<<<dim3((HW4 + 15) / 16, NIMG), 32, 0, stream>>>(
        x4, ws + OFF_M4, ws + OFF_SUM + 1 * NIMG, 28, 28, 2, 25, 2, 25);
    channel_mean_wmma_kernel<<<dim3((HW5 + 15) / 16, NIMG), 32, 0, stream>>>(
        x5, ws + OFF_M5, ws + OFF_SUM + 2 * NIMG, 14, 14, 1, 12, 1, 12);

    // per-(image,level) NMS
    nms_level_kernel<<<dim3(NIMG, 3), 256, 0, stream>>>(ws);

    // union crop rect per image
    union_rect_kernel<<<1, NIMG, 0, stream>>>(ws);

    // bilinear crop-resize of x2 -> first output
    {
        size_t total = (size_t)NIMG * NCH * 56 * 56;
        resize_kernel<<<(unsigned)((total + 255) / 256), 256, 0, stream>>>(x2, ws, out);
    }

    // second NMS (scalar per image) + region means -> second output
    nms2_kernel<<<1, NIMG, 0, stream>>>(ws);
    crop_mean_kernel<<<NIMG, 256, 0, stream>>>(
        x5, ws, out + (size_t)NIMG * NCH * 56 * 56);
}
